// MultiHeadScaledProductAttention_80470507258062
// MI455X (gfx1250) — matrix operand order repaired and checked by reading, not compiled, not run on hardware
//
#include <hip/hip_runtime.h>
#include <hip/hip_bf16.h>

typedef unsigned short u16;
typedef __attribute__((ext_vector_type(16))) __bf16 bf16x16;
typedef __attribute__((ext_vector_type(8)))  float  f32x8;
typedef __attribute__((ext_vector_type(4)))  unsigned int u32x4;
typedef __attribute__((ext_vector_type(4)))  float  f32x4;

union BFPack { u32x4 u[2]; bf16x16 v; };

#define WMMA_BF16(A, B, C) \
  __builtin_amdgcn_wmma_f32_16x16x32_bf16(false, (A), false, (B), (short)0, (C), false, false)

__device__ __forceinline__ u16 f2bf(float f) {
  unsigned int x = __float_as_uint(f);
  x += 0x7FFFu + ((x >> 16) & 1u);   // round-to-nearest-even
  return (u16)(x >> 16);
}

#define BB   8
#define TQS  1024
#define TMS  1024
#define DD   512
#define AT   512
#define HH   8
#define HD   64
#define NEGINF (-4294967295.0f)  // -2^32 + 1

// ---------------- kernel 1: f32 -> bf16 activation conversion ----------------
__global__ void cvt_x_kernel(const float* __restrict__ xin, const float* __restrict__ xmem,
                             u16* __restrict__ oin, u16* __restrict__ omem) {
  int i = blockIdx.x * 256 + threadIdx.x;           // 4,194,304 elements each
  oin[i]  = f2bf(xin[i]);
  omem[i] = f2bf(xmem[i]);
}

// ---------------- kernel 2: weights f32 (K,N) -> bf16 transposed (N,K) -------
__global__ void cvt_w_kernel(const float* __restrict__ Wq, const float* __restrict__ Wk,
                             const float* __restrict__ Wv, u16* __restrict__ WtA) {
  int i   = blockIdx.x * 256 + threadIdx.x;         // 3 * 512 * 512
  int w   = i >> 18;
  int rem = i & 262143;
  int n   = rem >> 9;
  int k   = rem & 511;
  const float* W = (w == 0) ? Wq : ((w == 1) ? Wk : Wv);
  WtA[i] = f2bf(W[k * AT + n]);
}

// ---------------- kernel 3: QKV projection GEMM via WMMA ---------------------
// One wave per 16x16 output tile; M = B*T = 8192, N = 512, K = 512.
__global__ void proj_wmma_kernel(const u16* __restrict__ Xq, const u16* __restrict__ Xm,
                                 const u16* __restrict__ WtA,
                                 u16* __restrict__ Qb, u16* __restrict__ Kb,
                                 u16* __restrict__ Vt) {
  int wave = threadIdx.x >> 5;
  int lane = threadIdx.x & 31;
  int tile = blockIdx.x * 4 + wave;                 // 49152 tiles = 3 * 512 * 32
  int w    = tile >> 14;                            // which weight (0=Q,1=K,2=V)
  int rem  = tile & 16383;
  int mt   = rem >> 5;
  int nt   = rem & 31;

  const u16* X    = (w == 0) ? Xq : Xm;
  const u16* Wt   = WtA + (w << 18);
  int col = lane & 15;
  int hi  = lane >> 4;
  const u16* Arow = X  + (size_t)(mt * 16 + col) * DD;
  const u16* Brow = Wt + (size_t)(nt * 16 + col) * DD;
  int ka = hi * 8;                                  // A: k-base per lane half
  int kb = hi * 8;                                  // B: k-base per lane half

  f32x8 acc = {0.f, 0.f, 0.f, 0.f, 0.f, 0.f, 0.f, 0.f};
  for (int kk = 0; kk < DD; kk += 32) {
    BFPack a, b;
    a.u[0] = *(const u32x4*)(Arow + kk + ka);
    a.u[1] = *(const u32x4*)(Arow + kk + ka + 16);
    b.u[0] = *(const u32x4*)(Brow + kk + kb);
    b.u[1] = *(const u32x4*)(Brow + kk + kb + 16);
    acc = WMMA_BF16(a.v, b.v, acc);
  }

  int rb = hi * 8;
  for (int r = 0; r < 8; ++r) {
    int mm = mt * 16 + rb + r;
    int nn = nt * 16 + col;
    int bb = mm >> 10, t = mm & 1023;
    int h  = nn >> 6,  d = nn & 63;
    u16 v = f2bf(acc[r]);
    if (w == 0)      Qb[((size_t)(bb * HH + h) * TQS + t) * HD + d] = v;
    else if (w == 1) Kb[((size_t)(bb * HH + h) * TMS + t) * HD + d] = v;
    else             Vt[((size_t)(bb * HH + h) * HD + d) * TMS + t] = v;  // transposed
  }
}

// ---------------- kernel 4: fused attention (QK^T, softmax, PV) --------------
// Grid: (TQ/16, H, B); block = 128 threads (4 waves); dyn LDS = 96 KB.
__global__ void attn_wmma_kernel(const u16* __restrict__ Qb, const u16* __restrict__ Kb,
                                 const u16* __restrict__ Vt,
                                 const int* __restrict__ mlens, const int* __restrict__ qlens,
                                 float* __restrict__ ctx, float* __restrict__ align) {
  extern __shared__ char smem[];
  float* Lg = (float*)smem;                 // 16 x 1024 f32 logits/probs (64 KB)
  u16*   P  = (u16*)(smem + 65536);         // 16 x 1024 bf16 probs       (32 KB)

  int qt = blockIdx.x, h = blockIdx.y, b = blockIdx.z;
  int wave = threadIdx.x >> 5;
  int lane = threadIdx.x & 31;
  int col  = lane & 15;
  int hi   = lane >> 4;
  int q0   = qt * 16;
  int qlen = qlens[b];
  int mlen = mlens[b];
  int ka = hi * 8;
  int kb = hi * 8;

  // --- Phase 1: logits = (Q K^T) * 1/8, masked, -> LDS ---
  const u16* Qrow = Qb + ((size_t)(b * HH + h) * TQS + q0 + col) * HD;
  BFPack aq0, aq1;
  aq0.u[0] = *(const u32x4*)(Qrow + ka);
  aq0.u[1] = *(const u32x4*)(Qrow + ka + 16);
  aq1.u[0] = *(const u32x4*)(Qrow + 32 + ka);
  aq1.u[1] = *(const u32x4*)(Qrow + 32 + ka + 16);

  const u16* Kbase = Kb + (size_t)(b * HH + h) * TMS * HD;
  for (int mt = wave; mt < TMS / 16; mt += 4) {
    const u16* Krow = Kbase + (size_t)(mt * 16 + col) * HD;
    if (mt + 4 < TMS / 16)
      __builtin_prefetch((const void*)(Kbase + (size_t)((mt + 4) * 16 + col) * HD), 0, 0);
    BFPack b0, b1;
    b0.u[0] = *(const u32x4*)(Krow + kb);
    b0.u[1] = *(const u32x4*)(Krow + kb + 16);
    b1.u[0] = *(const u32x4*)(Krow + 32 + kb);
    b1.u[1] = *(const u32x4*)(Krow + 32 + kb + 16);
    f32x8 acc = {0.f, 0.f, 0.f, 0.f, 0.f, 0.f, 0.f, 0.f};
    acc = WMMA_BF16(aq0.v, b0.v, acc);
    acc = WMMA_BF16(aq1.v, b1.v, acc);
    for (int r = 0; r < 8; ++r) {
      int row = hi * 8 + r;
      int q = q0 + row;
      int m = mt * 16 + col;
      bool valid = (q < qlen) && (m < mlen) && (m <= q);   // causal + lengths
      Lg[row * TMS + m] = valid ? acc[r] * 0.125f : NEGINF;
    }
  }
  __syncthreads();

  // --- Phase 2: row softmax (f32 probs back to Lg, bf16 probs to P) ---
  for (int rr = 0; rr < 4; ++rr) {
    int row = wave + rr * 4;
    float* Lrow = Lg + row * TMS;
    float mx = -3.4e38f;
    for (int c = lane; c < TMS; c += 32) mx = fmaxf(mx, Lrow[c]);
    for (int s = 16; s > 0; s >>= 1) mx = fmaxf(mx, __shfl_xor(mx, s, 32));
    float sum = 0.f;
    for (int c = lane; c < TMS; c += 32) {
      float e = __expf(Lrow[c] - mx);
      Lrow[c] = e;
      sum += e;
    }
    for (int s = 16; s > 0; s >>= 1) sum += __shfl_xor(sum, s, 32);
    float inv = 1.0f / sum;
    u16* Prow = P + row * TMS;
    for (int c = lane; c < TMS; c += 32) {
      float p = Lrow[c] * inv;
      Lrow[c] = p;
      Prow[c] = f2bf(p);
    }
  }
  __syncthreads();

  // --- Phase 3: stream alignments tile (contiguous 64 KB) to global ---
  float* aout = align + ((size_t)((b * HH + h) * TQS + q0)) * TMS;
  const f32x4* src = (const f32x4*)Lg;
  f32x4* dst = (f32x4*)aout;
  for (int i = threadIdx.x; i < (16 * TMS) / 4; i += 128) dst[i] = src[i];

  // --- Phase 4: contexts = P @ V (each wave owns one 16-wide d-slice) ---
  int d0 = wave * 16;
  const u16* Vbase = Vt + ((size_t)(b * HH + h) * HD + d0 + col) * TMS;
  const u16* Prow2 = P + col * TMS;
  f32x8 acc2 = {0.f, 0.f, 0.f, 0.f, 0.f, 0.f, 0.f, 0.f};
  for (int mk = 0; mk < TMS; mk += 32) {
    BFPack a, bv;
    a.u[0]  = *(const u32x4*)(Prow2 + mk + ka);
    a.u[1]  = *(const u32x4*)(Prow2 + mk + ka + 16);
    bv.u[0] = *(const u32x4*)(Vbase + mk + kb);
    bv.u[1] = *(const u32x4*)(Vbase + mk + kb + 16);
    acc2 = WMMA_BF16(a.v, bv.v, acc2);
  }
  for (int r = 0; r < 8; ++r) {
    int q = q0 + hi * 8 + r;
    ctx[((size_t)(b * TQS + q)) * AT + h * HD + d0 + col] = acc2[r];
  }
}

// ---------------------------------------------------------------------------
extern "C" void kernel_launch(void* const* d_in, const int* in_sizes, int n_in,
                              void* d_out, int out_size, void* d_ws, size_t ws_size,
                              hipStream_t stream) {
  (void)in_sizes; (void)n_in; (void)out_size; (void)ws_size;
  const float* inputs = (const float*)d_in[0];
  const float* memory = (const float*)d_in[1];
  const float* Wq     = (const float*)d_in[2];
  const float* Wk     = (const float*)d_in[3];
  const float* Wv     = (const float*)d_in[4];
  const int*   mlens  = (const int*)d_in[5];
  const int*   qlens  = (const int*)d_in[6];

  float* ctx   = (float*)d_out;                          // (B, TQ, ATTN)
  float* align = ctx + (size_t)BB * TQS * AT;            // (B, H, TQ, TM)

  char* ws = (char*)d_ws;
  u16* Xin  = (u16*)(ws);                                //  8 MB: inputs bf16
  u16* Xmem = (u16*)(ws + 8388608);                      //  8 MB: memory bf16
  u16* WtA  = (u16*)(ws + 16777216);                     //  1.5 MB: Wq/Wk/Wv^T bf16
  u16* Qb   = (u16*)(ws + 18350080);                     //  8 MB: Q  (B,H,TQ,64)
  u16* Kb   = (u16*)(ws + 26738688);                     //  8 MB: K  (B,H,TM,64)
  u16* Vt   = (u16*)(ws + 35127296);                     //  8 MB: V^T (B,H,64,TM)

  cvt_x_kernel<<<16384, 256, 0, stream>>>(inputs, memory, Xin, Xmem);
  cvt_w_kernel<<<3072, 256, 0, stream>>>(Wq, Wk, Wv, WtA);
  proj_wmma_kernel<<<12288, 128, 0, stream>>>(Xin, Xmem, WtA, Qb, Kb, Vt);
  dim3 grid(TQS / 16, HH, BB);
  attn_wmma_kernel<<<grid, 128, 98304, stream>>>(Qb, Kb, Vt, mlens, qlens, ctx, align);
}
